// TransformerDecoder2dLayer_79190607004035
// MI455X (gfx1250) — matrix operand order repaired and checked by reading, not compiled, not run on hardware
//
#include <hip/hip_runtime.h>
#include <hip/hip_bf16.h>
#include <math.h>

typedef __bf16 bf16_t;
typedef __attribute__((ext_vector_type(16))) __bf16 v16bf;
typedef __attribute__((ext_vector_type(8)))  float  v8f;
typedef __attribute__((ext_vector_type(4)))  int    v4i;

#define EPSN 1e-6f

struct __align__(8) bf4 { bf16_t v[4]; };

// --------------------------------------------------------------------------
// CDNA5 async global->LDS staging (ASYNCcnt-tracked). Falls back to a
// synchronous VGPR round-trip if the builtin is unavailable.
// Builtin signature (from hipcc diagnostic): (v4i* gaddr, v4i* lds, Ii, Ii).
// --------------------------------------------------------------------------
#if __has_builtin(__builtin_amdgcn_global_load_async_to_lds_b128)
#define HAVE_ASYNC 1
#else
#define HAVE_ASYNC 0
#endif

typedef __attribute__((address_space(1))) v4i gv4i;  // global int4
typedef __attribute__((address_space(3))) v4i lv4i;  // LDS int4

__device__ __forceinline__ void cp64B(const bf16_t* g, bf16_t* l) {
#if HAVE_ASYNC
  gv4i* gp = (gv4i*)g;
  lv4i* lp = (lv4i*)l;
  // 64 contiguous bytes: 4x b128; imm offset applies to both global and LDS.
  __builtin_amdgcn_global_load_async_to_lds_b128(gp, lp, 0, 0);
  __builtin_amdgcn_global_load_async_to_lds_b128(gp, lp, 16, 0);
  __builtin_amdgcn_global_load_async_to_lds_b128(gp, lp, 32, 0);
  __builtin_amdgcn_global_load_async_to_lds_b128(gp, lp, 48, 0);
#else
  const uint4* gs = (const uint4*)g;
  uint4* ls = (uint4*)l;
  uint4 t0 = gs[0], t1 = gs[1], t2 = gs[2], t3 = gs[3];
  ls[0] = t0; ls[1] = t1; ls[2] = t2; ls[3] = t3;
#endif
}

template <int N>
__device__ __forceinline__ void wait_async() {
#if HAVE_ASYNC
#if __has_builtin(__builtin_amdgcn_s_wait_asynccnt)
  __builtin_amdgcn_s_wait_asynccnt(N);
#else
  asm volatile("s_wait_asynccnt %0" ::"n"(N) : "memory");
#endif
#endif
}

// ---------------------------------------------------------------------------
// Weight prep: Wt[n*K + k] = (bf16) W[k*N + n]   (transpose so GEMM B-tile is
// (N,K) row-major -> contiguous per-lane LDS fragment loads)
// ---------------------------------------------------------------------------
__global__ void k_transpose_bf16(const float* __restrict__ W, bf16_t* __restrict__ Wt,
                                 int K, int N) {
  size_t i = (size_t)blockIdx.x * 256 + threadIdx.x;
  size_t total = (size_t)K * N;
  if (i >= total) return;
  int n = (int)(i / K);
  int k = (int)(i % K);
  Wt[i] = (bf16_t)W[(size_t)k * N + n];
}

// ---------------------------------------------------------------------------
// WMMA GEMM: C[M,N] = A[M,K](bf16,row-major) * Wt[N,K](bf16)^T + bias
// Block tile 128x128, BK=64, double-buffered async LDS staging.
// 256 threads = 8 wave32s; wave tile 32x64 -> 8 f32 accumulators.
// 16 v_wmma per barrier pair. M%128==0, N%128==0, K%64==0 (guaranteed).
// ---------------------------------------------------------------------------
template <int OUT_BF16, int DO_GELU>
__global__ __launch_bounds__(256) void k_gemm(const bf16_t* __restrict__ A,
                                              const bf16_t* __restrict__ Wt,
                                              const float* __restrict__ bias,
                                              void* __restrict__ Cout,
                                              int M, int N, int K) {
  __shared__ bf16_t As[2][128 * 64];
  __shared__ bf16_t Bs[2][128 * 64];
  const int tid  = threadIdx.x;
  const int lane = tid & 31;
  const int wave = tid >> 5;
  const int lh   = lane & 15;   // column / row-in-tile
  const int sel  = lane >> 4;   // lane-half selector
  const int wr   = wave >> 1;   // wave row group (0..3) -> 32 rows
  const int wc   = wave & 1;    // wave col group (0..1) -> 64 cols
  const int rowBlk = blockIdx.y * 128;
  const int colBlk = blockIdx.x * 128;
  const int crow = tid >> 1;        // copy row 0..127
  const int ckc  = (tid & 1) * 32;  // copy K base 0 or 32 (64B per thread)

  const bf16_t* gA = A  + (size_t)(rowBlk + crow) * K + ckc;
  const bf16_t* gB = Wt + (size_t)(colBlk + crow) * K + ckc;

  v8f acc[2][4];
  for (int i = 0; i < 2; ++i)
    for (int j = 0; j < 4; ++j)
      acc[i][j] = (v8f){0.f, 0.f, 0.f, 0.f, 0.f, 0.f, 0.f, 0.f};

  const int nT = K >> 6;
  // prologue: stage tile 0 into buffer 0 (8 async b128 per thread total)
  cp64B(gA, &As[0][crow * 64 + ckc]);
  cp64B(gB, &Bs[0][crow * 64 + ckc]);

  int buf = 0;
  for (int t = 0; t < nT; ++t) {
    if (t + 1 < nT) {
      // stage tile t+1 into the other buffer; overlaps with this tile's WMMAs
      cp64B(gA + (size_t)(t + 1) * 64, &As[buf ^ 1][crow * 64 + ckc]);
      cp64B(gB + (size_t)(t + 1) * 64, &Bs[buf ^ 1][crow * 64 + ckc]);
      wait_async<8>();  // drain tile t's 8 copies (in-order completion)
    } else {
      wait_async<0>();
    }
    __syncthreads();  // all waves' copies of buffer `buf` visible

    const bf16_t* Ab = &As[buf][0];
    const bf16_t* Bb = &Bs[buf][0];
#pragma unroll
    for (int ks = 0; ks < 64; ks += 32) {
      // A fragments (16x32 bf16): lanes 0-15 hold K {0..7,16..23} of row lh,
      // lanes 16-31 hold K {8..15,24..31}.
      v16bf afrag[2];
#pragma unroll
      for (int i = 0; i < 2; ++i) {
        int r = wr * 32 + i * 16 + lh;
        union { v16bf v; uint4 q[2]; } u;
        u.q[0] = *(const uint4*)(Ab + r * 64 + ks + sel * 8);
        u.q[1] = *(const uint4*)(Ab + r * 64 + ks + 16 + sel * 8);
        afrag[i] = u.v;
      }
      // B fragments (32x16 bf16): lane = column lh; lanes 0-15 hold K
      // {0..7,16..23}, lanes 16-31 hold K {8..15,24..31}.
      v16bf bfrag[4];
#pragma unroll
      for (int j = 0; j < 4; ++j) {
        int n = wc * 64 + j * 16 + lh;
        union { v16bf v; uint4 q[2]; } u;
        u.q[0] = *(const uint4*)(Bb + n * 64 + ks + sel * 8);
        u.q[1] = *(const uint4*)(Bb + n * 64 + ks + 16 + sel * 8);
        bfrag[j] = u.v;
      }
#pragma unroll
      for (int i = 0; i < 2; ++i)
#pragma unroll
        for (int j = 0; j < 4; ++j)
          acc[i][j] = __builtin_amdgcn_wmma_f32_16x16x32_bf16(
              false, afrag[i], false, bfrag[j], (short)0, acc[i][j], false, false);
    }
    __syncthreads();  // all reads done before next iteration overwrites buf
    buf ^= 1;
  }

  // Epilogue: C layout = VGPR r -> row (r + sel*8), lane lh -> column.
  for (int i = 0; i < 2; ++i)
    for (int j = 0; j < 4; ++j) {
      int rbase = rowBlk + wr * 32 + i * 16 + sel * 8;
      int col   = colBlk + wc * 64 + j * 16 + lh;
      float bv = bias[col];
      for (int r = 0; r < 8; ++r) {
        float v = acc[i][j][r] + bv;
        if (DO_GELU) v = 0.5f * v * (1.f + erff(v * 0.70710678118654752f));
        size_t idx = (size_t)(rbase + r) * N + col;
        if (OUT_BF16) ((bf16_t*)Cout)[idx] = (bf16_t)v;
        else          ((float*)Cout)[idx]  = v;
      }
    }
}

// ---------------------------------------------------------------------------
// Spatial concat: rows (f,b,s) s=0 -> timestep emb, s>=1 -> x[f,b,s-1]; bf16.
// ---------------------------------------------------------------------------
__global__ void k_concat_spatial(const float* __restrict__ x, const float* __restrict__ ts,
                                 bf16_t* __restrict__ out, int RS) {
  size_t c = (size_t)blockIdx.x * 256 + threadIdx.x;
  if (c >= (size_t)RS * 128) return;
  int m  = (int)(c >> 7);
  int d4 = (int)(c & 127) * 4;
  int s  = m % 23;
  int fb = m / 23;
  int b  = fb % 64;
  int f  = fb / 64;
  const float* src = (s == 0)
      ? ts + (size_t)b * 512 + d4
      : x + ((((size_t)f * 64 + b) * 22 + (s - 1)) * 512) + d4;
  float4 val = *(const float4*)src;
  bf4 o;
  o.v[0] = (bf16_t)val.x; o.v[1] = (bf16_t)val.y;
  o.v[2] = (bf16_t)val.z; o.v[3] = (bf16_t)val.w;
  *(bf4*)(out + (size_t)m * 512 + d4) = o;
}

// Temporal concat: rows (j,b,s); s=0 -> ts emb, s>=1 -> xn[f=s-1, b, j] (folds
// the swapaxes(0,2) permutation into the gather).
__global__ void k_concat_temporal(const float* __restrict__ xn, const float* __restrict__ ts,
                                  bf16_t* __restrict__ out, int RT) {
  size_t c = (size_t)blockIdx.x * 256 + threadIdx.x;
  if (c >= (size_t)RT * 128) return;
  int m  = (int)(c >> 7);
  int d4 = (int)(c & 127) * 4;
  int s  = m % 121;
  int jb = m / 121;
  int b  = jb % 64;
  int j  = jb / 64;
  const float* src = (s == 0)
      ? ts + (size_t)b * 512 + d4
      : xn + ((((size_t)(s - 1) * 64 + b) * 22 + j) * 512) + d4;
  float4 val = *(const float4*)src;
  bf4 o;
  o.v[0] = (bf16_t)val.x; o.v[1] = (bf16_t)val.y;
  o.v[2] = (bf16_t)val.z; o.v[3] = (bf16_t)val.w;
  *(bf4*)(out + (size_t)m * 512 + d4) = o;
}

// xcur = x + proj (same layout)
__global__ void k_add(const float* __restrict__ a, const float* __restrict__ b,
                      float* __restrict__ o, int rows) {
  size_t c = (size_t)blockIdx.x * 256 + threadIdx.x;
  if (c >= (size_t)rows * 128) return;
  size_t i = c * 4;
  float4 va = *(const float4*)(a + i);
  float4 vb = *(const float4*)(b + i);
  float4 vo = {va.x + vb.x, va.y + vb.y, va.z + vb.z, va.w + vb.w};
  *(float4*)(o + i) = vo;
}

// Temporal residual: out rows in (j,b,f) order = xn[f,b,j] + proj[(j,b,f)]
__global__ void k_add_perm(const float* __restrict__ xn, const float* __restrict__ proj,
                           float* __restrict__ o, int rows) {
  size_t c = (size_t)blockIdx.x * 256 + threadIdx.x;
  if (c >= (size_t)rows * 128) return;
  int m  = (int)(c >> 7);
  int d4 = (int)(c & 127) * 4;
  int f  = m % 120;
  int jb = m / 120;
  int b  = jb % 64;
  int j  = jb / 64;
  size_t src = ((((size_t)f * 64 + b) * 22 + j) * 512) + d4;
  float4 va = *(const float4*)(xn + src);
  float4 vb = *(const float4*)(proj + (size_t)m * 512 + d4);
  float4 vo = {va.x + vb.x, va.y + vb.y, va.z + vb.z, va.w + vb.w};
  *(float4*)(o + (size_t)m * 512 + d4) = vo;
}

// FFN prep: gather xn2 (j,b,f order) into (f,b,j) order; emit bf16 GEMM input
// plus an f32 copy kept for the final residual.
__global__ void k_prep_ffn(const float* __restrict__ xn2, bf16_t* __restrict__ outb,
                           float* __restrict__ outf, int rows) {
  size_t c = (size_t)blockIdx.x * 256 + threadIdx.x;
  if (c >= (size_t)rows * 128) return;
  int m  = (int)(c >> 7);
  int d4 = (int)(c & 127) * 4;
  int j  = m % 22;
  int fb = m / 22;
  int b  = fb % 64;
  int f  = fb / 64;
  size_t src = ((((size_t)j * 64 + b) * 120 + f) * 512) + d4;
  float4 v = *(const float4*)(xn2 + src);
  *(float4*)(outf + (size_t)m * 512 + d4) = v;
  bf4 o;
  o.v[0] = (bf16_t)v.x; o.v[1] = (bf16_t)v.y;
  o.v[2] = (bf16_t)v.z; o.v[3] = (bf16_t)v.w;
  *(bf4*)(outb + (size_t)m * 512 + d4) = o;
}

// Custom norm: mean, ddof=1 std, eps added to std. One wave per 512-wide row.
// Optional second operand Y fuses the pre-norm residual add.
__global__ __launch_bounds__(256) void k_norm(const float* __restrict__ X,
                                              const float* __restrict__ Y,
                                              const float* __restrict__ ga,
                                              const float* __restrict__ gb,
                                              float* __restrict__ out, int rows) {
  int wave = threadIdx.x >> 5, lane = threadIdx.x & 31;
  int row = blockIdx.x * 8 + wave;
  if (row >= rows) return;
  const float* xr = X + (size_t)row * 512;
  const float* yr = Y ? Y + (size_t)row * 512 : nullptr;
  float v[16];
  float sum = 0.f, sq = 0.f;
  for (int k = 0; k < 16; ++k) {
    int d = lane + 32 * k;
    float t = xr[d];
    if (Y) t += yr[d];
    v[k] = t;
    sum += t;
    sq += t * t;
  }
  for (int o = 16; o > 0; o >>= 1) {
    sum += __shfl_xor(sum, o, 32);
    sq  += __shfl_xor(sq,  o, 32);
  }
  float mean = sum * (1.f / 512.f);
  float var  = (sq - 512.f * mean * mean) * (1.f / 511.f);  // ddof=1
  float sd   = sqrtf(fmaxf(var, 0.f));
  float inv  = 1.f / (sd + EPSN);
  float* orow = out + (size_t)row * 512;
  for (int k = 0; k < 16; ++k) {
    int d = lane + 32 * k;
    orow[d] = ga[d] * (v[k] - mean) * inv + gb[d];
  }
}

// ---------------------------------------------------------------------------
// Attention (unmasked; masks are all-ones): one block per (group, head).
// K/V head slices staged in dynamic LDS; wave-per-query softmax via wave32
// __shfl_xor reductions. Query row 0 (timestep emb) is skipped: the reference
// slices [:, :, 1:, :] before the row-wise output projection.
// ---------------------------------------------------------------------------
__global__ __launch_bounds__(256) void k_attn(const bf16_t* __restrict__ Qp,
                                              const bf16_t* __restrict__ Kp,
                                              const bf16_t* __restrict__ Vp,
                                              bf16_t* __restrict__ Op, int S) {
  extern __shared__ char smem[];
  bf16_t* sK = (bf16_t*)smem;                   // S*128 bf16
  bf16_t* sV = sK + (size_t)S * 128;            // S*128 bf16
  float*  qsh = (float*)(sV + (size_t)S * 128); // 8 waves * 128
  float*  ssc = qsh + 8 * 128;                  // 8 waves * 128 (S <= 121)
  const int g = blockIdx.x >> 2;
  const int h = blockIdx.x & 3;
  const size_t base = (size_t)g * S * 512 + (size_t)h * 128;
  for (int c = threadIdx.x; c < S * 16; c += 256) {
    int s = c >> 4, q8 = (c & 15) * 8;
    *(uint4*)(sK + s * 128 + q8) = *(const uint4*)(Kp + base + (size_t)s * 512 + q8);
    *(uint4*)(sV + s * 128 + q8) = *(const uint4*)(Vp + base + (size_t)s * 512 + q8);
  }
  __syncthreads();
  const int wave = threadIdx.x >> 5, lane = threadIdx.x & 31;
  const float scale = 0.08838834764831845f;  // 1/sqrt(128)
  float* qw = qsh + wave * 128;
  float* sw = ssc + wave * 128;
  for (int q = 1 + wave; q < S; q += 8) {
    const bf16_t* qr = Qp + base + (size_t)q * 512;
    for (int k = 0; k < 4; ++k) {
      int d = lane + 32 * k;
      qw[d] = (float)qr[d];
    }
    // scores: lane j handles keys j, j+32, ...
    float mymax = -1e30f;
    for (int j = lane; j < S; j += 32) {
      const bf16_t* kr = sK + j * 128;
      float dot = 0.f;
      for (int d = 0; d < 128; ++d) dot += qw[d] * (float)kr[d];
      float sc = dot * scale;
      sw[j] = sc;
      mymax = fmaxf(mymax, sc);
    }
    for (int o = 16; o > 0; o >>= 1) mymax = fmaxf(mymax, __shfl_xor(mymax, o, 32));
    float mysum = 0.f;
    for (int j = lane; j < S; j += 32) {
      float e = __expf(sw[j] - mymax);
      sw[j] = e;
      mysum += e;
    }
    for (int o = 16; o > 0; o >>= 1) mysum += __shfl_xor(mysum, o, 32);
    float inv = 1.f / mysum;
    // output: lane owns dims {lane, lane+32, lane+64, lane+96}
    float a0 = 0.f, a1 = 0.f, a2 = 0.f, a3 = 0.f;
    for (int j = 0; j < S; ++j) {
      float p = sw[j];
      const bf16_t* vr = sV + j * 128;
      a0 += p * (float)vr[lane];
      a1 += p * (float)vr[lane + 32];
      a2 += p * (float)vr[lane + 64];
      a3 += p * (float)vr[lane + 96];
    }
    bf16_t* orow = Op + ((size_t)g * (S - 1) + (q - 1)) * 512 + (size_t)h * 128;
    orow[lane]      = (bf16_t)(a0 * inv);
    orow[lane + 32] = (bf16_t)(a1 * inv);
    orow[lane + 64] = (bf16_t)(a2 * inv);
    orow[lane + 96] = (bf16_t)(a3 * inv);
  }
}

// ---------------------------------------------------------------------------
extern "C" void kernel_launch(void* const* d_in, const int* in_sizes, int n_in,
                              void* d_out, int out_size, void* d_ws, size_t ws_size,
                              hipStream_t stream) {
  (void)in_sizes; (void)n_in; (void)out_size; (void)ws_size;
  const int F = 120, Bb = 64, Jj = 22, Dd = 512, Hh = 4, DFFc = 2048;
  const int R  = F * Bb * Jj;        // 168960  (multiple of 128)
  const int RS = F * Bb * (Jj + 1);  // 176640  (multiple of 128)
  const int RT = Jj * Bb * (F + 1);  // 170368  (multiple of 128)

  const float* x   = (const float*)d_in[0];
  const float* ts  = (const float*)d_in[1];
  const float* sWq = (const float*)d_in[5],  *sbq = (const float*)d_in[6];
  const float* sWk = (const float*)d_in[7],  *sbk = (const float*)d_in[8];
  const float* sWv = (const float*)d_in[9],  *sbv = (const float*)d_in[10];
  const float* sWo = (const float*)d_in[11], *sbo = (const float*)d_in[12];
  const float* tWq = (const float*)d_in[13], *tbq = (const float*)d_in[14];
  const float* tWk = (const float*)d_in[15], *tbk = (const float*)d_in[16];
  const float* tWv = (const float*)d_in[17], *tbv = (const float*)d_in[18];
  const float* tWo = (const float*)d_in[19], *tbo = (const float*)d_in[20];
  const float* ffW1 = (const float*)d_in[21], *ffb1 = (const float*)d_in[22];
  const float* ffW2 = (const float*)d_in[23], *ffb2 = (const float*)d_in[24];
  const float* n1a = (const float*)d_in[25], *n1b = (const float*)d_in[26];
  const float* n2a = (const float*)d_in[27], *n2b = (const float*)d_in[28];
  const float* n3a = (const float*)d_in[29], *n3b = (const float*)d_in[30];

  // ---- workspace carve (256B aligned) ----
  char* W = (char*)d_ws;
  size_t off = 0;
  auto carve = [&](size_t bytes) -> size_t {
    size_t o = off;
    off += (bytes + 255) & ~(size_t)255;
    return o;
  };
  bf16_t* wT[10];
  const size_t DDb = (size_t)Dd * Dd * 2;
  for (int i = 0; i < 8; ++i) wT[i] = (bf16_t*)(W + carve(DDb));
  wT[8] = (bf16_t*)(W + carve((size_t)Dd * DFFc * 2));  // ffW1^T (2048,512)
  wT[9] = (bf16_t*)(W + carve((size_t)Dd * DFFc * 2));  // ffW2^T (512,2048)
  bf16_t* catA  = (bf16_t*)(W + carve((size_t)RS * Dd * 2));
  bf16_t* Qb    = (bf16_t*)(W + carve((size_t)RS * Dd * 2));
  bf16_t* Kb    = (bf16_t*)(W + carve((size_t)RS * Dd * 2));
  bf16_t* Vb    = (bf16_t*)(W + carve((size_t)RS * Dd * 2));
  bf16_t* attnO = (bf16_t*)(W + carve((size_t)R * Dd * 2));
  float*  proj  = (float*)(W + carve((size_t)R * Dd * 4));
  float*  xcur  = (float*)(W + carve((size_t)R * Dd * 4));
  float*  xn    = (float*)(W + carve((size_t)R * Dd * 4));
  // FFN hidden (R x 2048 bf16, ~660MB) reuses the then-dead Q/K/V/attnO span
  bf16_t* h1 = Qb;

  // ---- stage 0: transpose + bf16-convert all weights ----
  const float* wsrc[10] = {sWq, sWk, sWv, sWo, tWq, tWk, tWv, tWo, ffW1, ffW2};
  const int wK[10] = {Dd, Dd, Dd, Dd, Dd, Dd, Dd, Dd, Dd, DFFc};
  const int wN[10] = {Dd, Dd, Dd, Dd, Dd, Dd, Dd, Dd, DFFc, Dd};
  for (int i = 0; i < 10; ++i) {
    size_t tot = (size_t)wK[i] * wN[i];
    k_transpose_bf16<<<dim3((unsigned)((tot + 255) / 256)), 256, 0, stream>>>(
        wsrc[i], wT[i], wK[i], wN[i]);
  }

  const unsigned gEltS = (unsigned)(((size_t)RS * 128 + 255) / 256);
  const unsigned gEltT = (unsigned)(((size_t)RT * 128 + 255) / 256);
  const unsigned gEltR = (unsigned)(((size_t)R * 128 + 255) / 256);

  // ---- stage 1: spatial attention ----
  k_concat_spatial<<<dim3(gEltS), 256, 0, stream>>>(x, ts, catA, RS);
  dim3 gS(512 / 128, RS / 128);
  k_gemm<1, 0><<<gS, 256, 0, stream>>>(catA, wT[0], sbq, Qb, RS, 512, 512);
  k_gemm<1, 0><<<gS, 256, 0, stream>>>(catA, wT[1], sbk, Kb, RS, 512, 512);
  k_gemm<1, 0><<<gS, 256, 0, stream>>>(catA, wT[2], sbv, Vb, RS, 512, 512);
  size_t shm_s = (size_t)23 * 512 + 8192;   // K+V tiles + q/score scratch
  k_attn<<<dim3(F * Bb * Hh), 256, shm_s, stream>>>(Qb, Kb, Vb, attnO, 23);
  dim3 gP(512 / 128, R / 128);
  k_gemm<0, 0><<<gP, 256, 0, stream>>>(attnO, wT[3], sbo, proj, R, 512, 512);
  k_add<<<dim3(gEltR), 256, 0, stream>>>(x, proj, xcur, R);
  k_norm<<<dim3(R / 8), 256, 0, stream>>>(xcur, nullptr, n1a, n1b, xn, R);

  // ---- stage 2: temporal attention (permutation folded into gathers) ----
  k_concat_temporal<<<dim3(gEltT), 256, 0, stream>>>(xn, ts, catA, RT);
  dim3 gT(512 / 128, RT / 128);
  k_gemm<1, 0><<<gT, 256, 0, stream>>>(catA, wT[4], tbq, Qb, RT, 512, 512);
  k_gemm<1, 0><<<gT, 256, 0, stream>>>(catA, wT[5], tbk, Kb, RT, 512, 512);
  k_gemm<1, 0><<<gT, 256, 0, stream>>>(catA, wT[6], tbv, Vb, RT, 512, 512);
  size_t shm_t = (size_t)121 * 512 + 8192;  // ~70KB dynamic LDS (WGP has 320KB)
  k_attn<<<dim3(Jj * Bb * Hh), 256, shm_t, stream>>>(Qb, Kb, Vb, attnO, 121);
  k_gemm<0, 0><<<gP, 256, 0, stream>>>(attnO, wT[7], tbo, proj, R, 512, 512);
  k_add_perm<<<dim3(gEltR), 256, 0, stream>>>(xn, proj, xcur, R);
  k_norm<<<dim3(R / 8), 256, 0, stream>>>(xcur, nullptr, n2a, n2b, xn, R);

  // ---- stage 3: FFN (gather back to (F,B,J) order) + final add-norm ----
  k_prep_ffn<<<dim3(gEltR), 256, 0, stream>>>(xn, catA, xcur, R);
  dim3 gF1(DFFc / 128, R / 128);
  k_gemm<1, 1><<<gF1, 256, 0, stream>>>(catA, wT[8], ffb1, h1, R, DFFc, 512);
  dim3 gF2(512 / 128, R / 128);
  k_gemm<0, 0><<<gF2, 256, 0, stream>>>(h1, wT[9], ffb2, proj, R, 512, DFFc);
  k_norm<<<dim3(R / 8), 256, 0, stream>>>(xcur, proj, n3a, n3b, (float*)d_out, R);
}
